// KernelNN_4827543241025
// MI455X (gfx1250) — matrix operand order repaired and checked by reading, not compiled, not run on hardware
//
#include <hip/hip_runtime.h>
#include <hip/hip_bf16.h>
#include <stdint.h>

typedef __attribute__((ext_vector_type(16))) _Float16 v16h;
typedef __attribute__((ext_vector_type(8)))  float    v8f;

union FragU { v16h h; unsigned int u[8]; };

#define WIDTH     64
#define KERW      1024
#define KOUT      4096   // WIDTH*WIDTH

// ---------------------------------------------------------------------------
// CDNA5 async global->LDS copy (16B per lane), tracked by ASYNCcnt.
// Generic LDS pointer low 32 bits == LDS byte offset.
// ---------------------------------------------------------------------------
__device__ __forceinline__ void async_b128(const _Float16* g, _Float16* l) {
    unsigned loff = (unsigned)(size_t)l;
    unsigned long long ga = (unsigned long long)(size_t)g;
    asm volatile("global_load_async_to_lds_b128 %0, %1, off"
                 :: "v"(loff), "v"(ga) : "memory");
}
__device__ __forceinline__ void wait_async0() {
    asm volatile("s_wait_asynccnt 0x0" ::: "memory");
}

// ---------------------------------------------------------------------------
// fp32 -> fp16 weight conversion
// ---------------------------------------------------------------------------
__global__ void cvt_f32_f16(const float* __restrict__ in, _Float16* __restrict__ out, int n) {
    int i = blockIdx.x * 256 + threadIdx.x;
    if (i < n) out[i] = (_Float16)in[i];
}

// ---------------------------------------------------------------------------
// Transpose + convert: wt[n][k] = (f16) w[k][n].  32x32 tiles via LDS.
// ---------------------------------------------------------------------------
__global__ __launch_bounds__(256) void transpose_f32_f16(
        const float* __restrict__ w, _Float16* __restrict__ wt, int K, int N) {
    __shared__ float t[32][33];
    int kb = blockIdx.x * 32, nb = blockIdx.y * 32;
    int tx = threadIdx.x & 31, ty = threadIdx.x >> 5;   // 32 x 8
    #pragma unroll
    for (int i = ty; i < 32; i += 8)
        t[i][tx] = w[(size_t)(kb + i) * N + nb + tx];
    __syncthreads();
    #pragma unroll
    for (int i = ty; i < 32; i += 8)
        wt[(size_t)(nb + i) * K + kb + tx] = (_Float16)t[tx][i];
}

// ---------------------------------------------------------------------------
// h0 = x @ fc1_w + fc1_b   (IN_WIDTH == 1)
// ---------------------------------------------------------------------------
__global__ void fc1_kernel(const float* __restrict__ x, const float* __restrict__ w,
                           const float* __restrict__ b, float* __restrict__ h, int N) {
    int i = blockIdx.x * 256 + threadIdx.x;
    if (i >= N * WIDTH) return;
    int n = i >> 6, o = i & 63;
    h[i] = x[n] * w[o] + b[o];
}

// ---------------------------------------------------------------------------
// e1 = relu(edge_attr @ k1_w + k1_b)   [E,6]x[6,1024], output f16
// ---------------------------------------------------------------------------
__global__ __launch_bounds__(256) void edge_mlp1(const float* __restrict__ attr,
                                                 const _Float16* __restrict__ w,
                                                 const float* __restrict__ b,
                                                 _Float16* __restrict__ out, int E) {
    int e = blockIdx.x;
    if (e >= E) return;
    float a0 = attr[e*6+0], a1 = attr[e*6+1], a2 = attr[e*6+2];
    float a3 = attr[e*6+3], a4 = attr[e*6+4], a5 = attr[e*6+5];
    #pragma unroll
    for (int rep = 0; rep < 4; ++rep) {
        int o = threadIdx.x + rep * 256;
        float acc = b[o];
        acc += a0 * (float)w[0*KERW + o];
        acc += a1 * (float)w[1*KERW + o];
        acc += a2 * (float)w[2*KERW + o];
        acc += a3 * (float)w[3*KERW + o];
        acc += a4 * (float)w[4*KERW + o];
        acc += a5 * (float)w[5*KERW + o];
        out[(size_t)e*KERW + o] = (_Float16)fmaxf(acc, 0.f);
    }
}

// ---------------------------------------------------------------------------
// WMMA GEMM: C[M,N] = act(A[M,K] @ B[K,N] + bias); B pre-transposed as Bt[N][K].
// f16 in, f32 acc, f16 out. Block tile 128x128, K-step 32, 8 waves (2x4),
// each wave: 64x32 sub-tile = 8 x v_wmma_f32_16x16x32_f16 per K-step.
// Double-buffered LDS via GLOBAL_LOAD_ASYNC_TO_LDS_B128 (ASYNCcnt).
// FULLM=true: branch-free staging/epilogue (all rows valid).
// FULLM=false: single ragged tail block with guarded synchronous A staging.
// ---------------------------------------------------------------------------
#define TM 128
#define TN 128
#define TK 32
#define TSTR 40   // halves per LDS row: 80B -> 16B aligned, bank-spread

template <bool FULLM>
__global__ __launch_bounds__(256) void gemm_f16_wmma(
        const _Float16* __restrict__ A,    // [M][K]
        const _Float16* __restrict__ Bt,   // [N][K]
        const float* __restrict__ bias, _Float16* __restrict__ C,
        int M, int N, int K, int m0, int relu) {
    __shared__ _Float16 ldsA[2][TM * TSTR];
    __shared__ _Float16 ldsB[2][TN * TSTR];

    const int tid   = threadIdx.x;
    const int lane  = tid & 31;
    const int wid   = tid >> 5;
    const int waveM = wid >> 2;            // 0..1 -> 64-row strip
    const int waveN = wid & 3;             // 0..3 -> 32-col strip
    const int blockM = m0 + blockIdx.x * TM;
    const int blockN = blockIdx.y * TN;

    // staging map: thread t -> row sr = t>>1, half-row offset sho = (t&1)*16
    const int sr  = tid >> 1;
    const int sho = (tid & 1) * 16;

    auto stage = [&](int buf, int kk) {
        const _Float16* ag = A + (size_t)(blockM + sr) * K + kk + sho;
        _Float16* al = &ldsA[buf][sr * TSTR + sho];
        if (FULLM) {
            async_b128(ag,     al);
            async_b128(ag + 8, al + 8);
        } else {
            uint4 v0 = make_uint4(0u,0u,0u,0u), v1 = make_uint4(0u,0u,0u,0u);
            if (blockM + sr < M) {
                v0 = *(const uint4*)ag;
                v1 = *(const uint4*)(ag + 8);
            }
            *(uint4*)al       = v0;
            *(uint4*)(al + 8) = v1;
        }
        const _Float16* bg = Bt + (size_t)(blockN + sr) * K + kk + sho;
        _Float16* bl = &ldsB[buf][sr * TSTR + sho];
        async_b128(bg,     bl);
        async_b128(bg + 8, bl + 8);
    };

    v8f acc[4][2] = {};

    stage(0, 0);                              // prefetch first K-step

    const int ksteps = K / TK;
    for (int ks = 0; ks < ksteps; ++ks) {
        const int buf = ks & 1;
        wait_async0();                        // own async fills done
        __syncthreads();                      // everyone's fills done; prev reads done
        if (ks + 1 < ksteps) stage(buf ^ 1, (ks + 1) * TK);

        // B fragments (loop-invariant this K-step): lanes 0-15 K=0..15, 16-31 K=16..31
        FragU bf[2];
        {
            int bn  = lane & 15;
            int bkb = (lane & 16) ? 16 : 0;
            #pragma unroll
            for (int j = 0; j < 2; ++j) {
                const unsigned int* rowp =
                    (const unsigned int*)(&ldsB[buf][(waveN * 32 + j * 16 + bn) * TSTR]);
                #pragma unroll
                for (int v = 0; v < 8; ++v)
                    bf[j].u[v] = rowp[(bkb >> 1) + v];
            }
        }
        // A fragments streamed over 4 row-groups; 2 WMMA each
        {
            int m   = lane & 15;
            int akb = (lane & 16) ? 16 : 0;
            #pragma unroll
            for (int i = 0; i < 4; ++i) {
                FragU af;
                const unsigned int* rowp =
                    (const unsigned int*)(&ldsA[buf][(waveM * 64 + i * 16 + m) * TSTR]);
                #pragma unroll
                for (int v = 0; v < 8; ++v) {
                    int k = akb + v * 2;
                    af.u[v] = rowp[k >> 1];
                }
                #pragma unroll
                for (int j = 0; j < 2; ++j)
                    acc[i][j] = __builtin_amdgcn_wmma_f32_16x16x32_f16(
                        false, af.h, false, bf[j].h, (short)0, acc[i][j], false, false);
            }
        }
        __syncthreads();
    }

    // epilogue: bias + optional relu, f16 store (C/D layout: VGPR r -> rows r / r+8)
    int n  = lane & 15;
    int mb = (lane & 16) ? 8 : 0;
    #pragma unroll
    for (int i = 0; i < 4; ++i) {
        #pragma unroll
        for (int j = 0; j < 2; ++j) {
            int col = blockN + waveN * 32 + j * 16 + n;
            float bv = bias[col];
            #pragma unroll
            for (int r = 0; r < 8; ++r) {
                int row = blockM + waveM * 64 + i * 16 + mb + r;
                if (FULLM || row < M) {
                    float val = acc[i][j][r] + bv;
                    if (relu) val = fmaxf(val, 0.f);
                    C[(size_t)row * N + col] = (_Float16)val;
                }
            }
        }
    }
}

// ---------------------------------------------------------------------------
// degree + inverse degree
// ---------------------------------------------------------------------------
__global__ void deg_kernel(const long long* __restrict__ dst, float* __restrict__ deg, int E) {
    int e = blockIdx.x * 256 + threadIdx.x;
    if (e < E) unsafeAtomicAdd(&deg[dst[e]], 1.0f);
}
__global__ void invdeg_kernel(const float* __restrict__ deg, float* __restrict__ inv, int N) {
    int n = blockIdx.x * 256 + threadIdx.x;
    if (n < N) { float d = deg[n]; inv[n] = (d > 0.f) ? (1.0f / d) : 0.f; }
}

// ---------------------------------------------------------------------------
// per-edge matvec msg = h[src] @ W_edge[e], scatter-add into agg[dst]
// ---------------------------------------------------------------------------
__global__ __launch_bounds__(256) void edge_message(
        const float* __restrict__ h, const _Float16* __restrict__ W,
        const long long* __restrict__ src, const long long* __restrict__ dst,
        float* __restrict__ agg, int E) {
    __shared__ float hs[8][WIDTH];
    int lane = threadIdx.x & 31, w = threadIdx.x >> 5;
    int e = blockIdx.x * 8 + w;
    long long d = 0;
    if (e < E) {
        long long s = src[e];
        d = dst[e];
        hs[w][lane]      = h[s * WIDTH + lane];
        hs[w][lane + 32] = h[s * WIDTH + lane + 32];
    }
    __syncthreads();
    if (e >= E) return;

    const _Float16* We = W + (size_t)e * KOUT;
    float a0 = 0.f, a1 = 0.f;
    #pragma unroll 4
    for (int c = 0; c < WIDTH; ++c) {
        float hv = hs[w][c];
        unsigned int p = *(const unsigned int*)(We + c * WIDTH + 2 * lane);
        _Float16 lo = ((const _Float16*)&p)[0];
        _Float16 hi = ((const _Float16*)&p)[1];
        a0 += hv * (float)lo;
        a1 += hv * (float)hi;
    }
    unsafeAtomicAdd(&agg[d * WIDTH + 2 * lane],     a0);
    unsafeAtomicAdd(&agg[d * WIDTH + 2 * lane + 1], a1);
}

// ---------------------------------------------------------------------------
// h_out = agg*inv_deg + h @ root + conv_b ; optional relu
// ---------------------------------------------------------------------------
__global__ __launch_bounds__(64) void node_update(
        const float* __restrict__ h, const float* __restrict__ agg,
        const float* __restrict__ invdeg, const float* __restrict__ root,
        const float* __restrict__ conv_b, float* __restrict__ hout,
        int N, int relu) {
    __shared__ float hn[WIDTH];
    int nId = blockIdx.x, o = threadIdx.x;
    hn[o] = h[nId * WIDTH + o];
    __syncthreads();
    float acc = agg[nId * WIDTH + o] * invdeg[nId] + conv_b[o];
    #pragma unroll
    for (int c = 0; c < WIDTH; ++c)
        acc += hn[c] * root[c * WIDTH + o];
    if (relu) acc = fmaxf(acc, 0.f);
    hout[nId * WIDTH + o] = acc;
}

// ---------------------------------------------------------------------------
// out = h @ fc2_w + fc2_b
// ---------------------------------------------------------------------------
__global__ void fc2_kernel(const float* __restrict__ h, const float* __restrict__ w,
                           const float* __restrict__ b, float* __restrict__ out, int N) {
    int n = blockIdx.x * 256 + threadIdx.x;
    if (n >= N) return;
    float acc = b[0];
    #pragma unroll
    for (int c = 0; c < WIDTH; ++c) acc += h[n * WIDTH + c] * w[c];
    out[n] = acc;
}

// ---------------------------------------------------------------------------
static inline uint8_t* carve(uint8_t** p, size_t bytes) {
    uint8_t* r = *p;
    *p += (bytes + 255) & ~(size_t)255;
    return r;
}

static inline void launch_gemm(const _Float16* A, const _Float16* Bt, const float* bias,
                               _Float16* C, int M, int N, int K, int relu,
                               hipStream_t stream) {
    int mfull = M / TM;                    // full 128-row blocks -> branch-free kernel
    if (mfull > 0) {
        dim3 g(mfull, N / TN);
        gemm_f16_wmma<true><<<g, 256, 0, stream>>>(A, Bt, bias, C, M, N, K, 0, relu);
    }
    if (M % TM) {                          // single ragged tail block
        dim3 g(1, N / TN);
        gemm_f16_wmma<false><<<g, 256, 0, stream>>>(A, Bt, bias, C, M, N, K, mfull * TM, relu);
    }
}

extern "C" void kernel_launch(void* const* d_in, const int* in_sizes, int n_in,
                              void* d_out, int out_size, void* d_ws, size_t ws_size,
                              hipStream_t stream) {
    const float*     x          = (const float*)d_in[0];
    const long long* edge_index = (const long long*)d_in[1];
    const float*     edge_attr  = (const float*)d_in[2];
    const float*     fc1_w      = (const float*)d_in[3];
    const float*     fc1_b      = (const float*)d_in[4];
    const float*     k1_w       = (const float*)d_in[5];
    const float*     k1_b       = (const float*)d_in[6];
    const float*     k2_w       = (const float*)d_in[7];
    const float*     k2_b       = (const float*)d_in[8];
    const float*     k3_w       = (const float*)d_in[9];
    const float*     k3_b       = (const float*)d_in[10];
    const float*     root       = (const float*)d_in[11];
    const float*     conv_b     = (const float*)d_in[12];
    const float*     fc2_w      = (const float*)d_in[13];
    const float*     fc2_b      = (const float*)d_in[14];

    const int N = in_sizes[0];          // x is [N,1]
    const int E = in_sizes[2] / 6;      // edge_attr is [E,6]
    const long long* src = edge_index;
    const long long* dst = edge_index + E;

    uint8_t* p = (uint8_t*)d_ws;
    _Float16* k1w16 = (_Float16*)carve(&p, (size_t)6 * KERW * 2);
    _Float16* k2t   = (_Float16*)carve(&p, (size_t)KERW * KERW * 2);   // [N][K]
    _Float16* k3t   = (_Float16*)carve(&p, (size_t)KERW * KOUT * 2);   // [N][K]
    _Float16* e1    = (_Float16*)carve(&p, (size_t)E * KERW * 2);
    _Float16* e2    = (_Float16*)carve(&p, (size_t)E * KERW * 2);
    _Float16* Wedge = (_Float16*)carve(&p, (size_t)E * KOUT * 2);
    float* hA     = (float*)carve(&p, (size_t)N * WIDTH * 4);
    float* hB     = (float*)carve(&p, (size_t)N * WIDTH * 4);
    float* agg    = (float*)carve(&p, (size_t)N * WIDTH * 4);
    float* deg    = (float*)carve(&p, (size_t)N * 4);
    float* invdeg = (float*)carve(&p, (size_t)N * 4);

    // 1) weight prep: k1 plain f16; k2/k3 transposed to [N][K] f16
    cvt_f32_f16<<<(6 * KERW + 255) / 256, 256, 0, stream>>>(k1_w, k1w16, 6 * KERW);
    {
        dim3 g2(KERW / 32, KERW / 32);
        transpose_f32_f16<<<g2, 256, 0, stream>>>(k2_w, k2t, KERW, KERW);
        dim3 g3(KERW / 32, KOUT / 32);
        transpose_f32_f16<<<g3, 256, 0, stream>>>(k3_w, k3t, KERW, KOUT);
    }

    // 2) node embedding
    fc1_kernel<<<(N * WIDTH + 255) / 256, 256, 0, stream>>>(x, fc1_w, fc1_b, hA, N);

    // 3) edge kernel-MLP (computed once; identical across the 6 layers)
    edge_mlp1<<<E, 256, 0, stream>>>(edge_attr, k1w16, k1_b, e1, E);
    launch_gemm(e1, k2t, k2_b, e2,    E, KERW, KERW, 1, stream);
    launch_gemm(e2, k3t, k3_b, Wedge, E, KOUT, KERW, 0, stream);

    // 4) degree / inverse degree for scatter-mean
    hipMemsetAsync(deg, 0, (size_t)N * 4, stream);
    deg_kernel<<<(E + 255) / 256, 256, 0, stream>>>(dst, deg, E);
    invdeg_kernel<<<(N + 255) / 256, 256, 0, stream>>>(deg, invdeg, N);

    // 5) 6 rounds of message passing
    float* hcur = hA;
    float* hnxt = hB;
    for (int k = 0; k < 6; ++k) {
        hipMemsetAsync(agg, 0, (size_t)N * WIDTH * 4, stream);
        edge_message<<<(E + 7) / 8, 256, 0, stream>>>(hcur, Wedge, src, dst, agg, E);
        node_update<<<N, 64, 0, stream>>>(hcur, agg, invdeg, root, conv_b, hnxt, N, (k < 5) ? 1 : 0);
        float* t = hcur; hcur = hnxt; hnxt = t;
    }

    // 6) readout
    fc2_kernel<<<(N + 255) / 256, 256, 0, stream>>>(hcur, fc2_w, fc2_b, (float*)d_out, N);
}
